// HyperbolicCentroidLoss_13305808683588
// MI455X (gfx1250) — matrix operand order repaired and checked by reading, not compiled, not run on hardware
//
#include <hip/hip_runtime.h>
#include <math.h>

#define AS1 __attribute__((address_space(1)))
#define AS3 __attribute__((address_space(3)))

typedef __attribute__((ext_vector_type(16))) __bf16 v16bf;
typedef __attribute__((ext_vector_type(8)))  float  v8f;
typedef __attribute__((ext_vector_type(4)))  int    v4i;

#if __has_builtin(__builtin_amdgcn_global_load_async_to_lds_b128) && \
    __has_builtin(__builtin_amdgcn_s_wait_asynccnt)
#define USE_ASYNC 1
#else
#define USE_ASYNC 0
#endif

// ---------------- problem geometry ----------------
#define NPTS     1048576u
#define DIMS     32
#define TILEPTS  256            // points staged per block tile (32 KB)
#define NT_TILES 4096u          // NPTS / TILEPTS
#define GRID_S   512
#define TPB_T    8              // NT_TILES / GRID_S

// ---------------- workspace layout (float offsets) ----------------
#define WS_SUMS4  0             // 81*32 fine-level segment sums
#define WS_CNT4   2592          // 81 fine-level counts
#define WS_DIST   2688          // 120 per-(level,seg) distance sums
#define WS_CENT   2816          // 120*32 centroids
#define WS_NY     6656          // 120 ||cent||^2
#define WS_CNTA   6784          // 120 per-(level,seg) counts
// total 6904 floats (~27 KB)

// ---------------- async staging helpers ----------------
__device__ __forceinline__ void cp16_async(const void* g, void* l) {
#if USE_ASYNC
  __builtin_amdgcn_global_load_async_to_lds_b128(
      (AS1 v4i*)(unsigned long long)(size_t)g,
      (AS3 v4i*)(unsigned)(size_t)l, 0, 0);
#else
  *(float4*)l = *(const float4*)g;
#endif
}

template <int N>
__device__ __forceinline__ void wait_async() {
#if USE_ASYNC
  __builtin_amdgcn_s_wait_asynccnt(N);
#endif
}

// stage one 256-point x 32-dim f32 tile (32 KB): 8 wave-wide b128 async ops
__device__ __forceinline__ void stage_tile(const float* zg, unsigned tile,
                                           float* lbuf, int tid) {
  const char* g = (const char*)zg + (size_t)tile * (TILEPTS * DIMS * 4) + (size_t)(tid * 16);
  char* l = (char*)lbuf + tid * 16;
#pragma unroll
  for (int j = 0; j < 8; ++j)
    cp16_async(g + (size_t)j * 4096, l + j * 4096);
}

// ---------------- kernel 0: zero the accumulator region ----------------
__global__ void k0_init(float* ws) {
  for (int i = threadIdx.x; i < 2816; i += blockDim.x) ws[i] = 0.0f;
}

// ---------------- kernel 1: projection + fine segment sums via WMMA ----------------
// One-hot bf16 A (16 segs x 32 pts, exact) times hi/lo-split bf16 B (32 pts x 16 dims),
// f32 accumulate: sums[seg][dim]. 6 seg-chunks x 2 dim-chunks x {hi,lo} = 24 WMMAs / 32 pts.
__global__ __launch_bounds__(256) void k1_sums(const float* __restrict__ z,
                                               const unsigned* __restrict__ bidx2,
                                               float* __restrict__ ws) {
  extern __shared__ float smem[];
  float* zbuf   = smem;                    // 2*256*32 = 16384 (double buffer)
  float* scArr  = smem + 16384;            // 256 per-point projection scales
  int*   segArr = (int*)(smem + 16640);    // 256 per-point fine segments
  float* cntB   = smem + 16896;            // 96 (81 used) block count bins
  float* binsS  = smem + 16992;            // 81*32 = 2592 block sum bins
  // total 19584 floats = 78336 bytes

  const int tid  = threadIdx.x;
  const int lane = tid & 31, wid = tid >> 5;
  const int half = lane >> 4, n16 = lane & 15;
  const int w32  = wid * 32;

  for (int i = tid; i < 96 + 2592; i += 256) cntB[i] = 0.0f;

  v8f acc[6][2];
#pragma unroll
  for (int c = 0; c < 6; ++c)
#pragma unroll
    for (int dc = 0; dc < 2; ++dc) acc[c][dc] = (v8f){0.f,0.f,0.f,0.f,0.f,0.f,0.f,0.f};

  const unsigned t0 = blockIdx.x;
  stage_tile(z, t0 + 0u * GRID_S, zbuf, tid);
  stage_tile(z, t0 + 1u * GRID_S, zbuf + 8192, tid);

#pragma unroll 1               // keep one iteration resident: stay under 256 VGPRs
  for (int i = 0; i < TPB_T; ++i) {
    if (i + 1 < TPB_T) wait_async<8>(); else wait_async<0>();
    __syncthreads();

    float* zb = zbuf + (i & 1) * 8192 + wid * (32 * DIMS);
    const unsigned tile = t0 + (unsigned)i * GRID_S;

    // --- per-point segment + Poincare projection scale (rotated, bank-clean) ---
    const unsigned p   = tile * TILEPTS + (unsigned)tid;
    const unsigned idx = bidx2[2u * p];             // low 32 bits of int64 index
    const int seg = (int)(idx / 243u);              // level-4 segment, 0..80
    float nx = 0.0f;
#pragma unroll
    for (int t = 0; t < 32; ++t) {
      int d = (lane + t) & 31;
      float v = zb[lane * DIMS + d];
      nx = fmaf(v, v, nx);
    }
    float nrm = sqrtf(nx);
    float s = (nrm > 0.95f) ? 0.95f / (nrm + 1e-12f) : 1.0f;
    segArr[w32 + lane] = seg;
    scArr[w32 + lane]  = s;
    atomicAdd(&cntB[seg], 1.0f);                    // ds_add_f32
    __builtin_amdgcn_wave_barrier();                // wave-internal DS ordering

    // --- per-lane gather of the k-specific seg/scale values this lane needs ---
    int   sk[16]; float sc[16];
#pragma unroll
    for (int j = 0; j < 8; ++j) {
      sk[j]     = segArr[w32 + half * 16 + j];       // A: K = 16h + j
      sk[8 + j] = segArr[w32 + half * 16 + 8 + j];   // A: K = 16h + 8 + j
    }
#pragma unroll
    for (int j = 0; j < 16; ++j) sc[j] = scArr[w32 + half * 16 + j];  // B: K = 16h + j

    // --- build B tiles (32 pts x 16 dims), hi/lo bf16 split of s*z ---
    union Bu { unsigned u[8]; v16bf v; } bH[2], bL[2];
#pragma unroll
    for (int dc = 0; dc < 2; ++dc) {
#pragma unroll
      for (int v = 0; v < 8; ++v) {
        int k0 = half * 16 + 2 * v;
        float z0 = zb[k0 * DIMS + dc * 16 + n16] * sc[2 * v];
        float z1 = zb[(k0 + 1) * DIMS + dc * 16 + n16] * sc[2 * v + 1];
        unsigned u0 = __float_as_uint(z0), u1 = __float_as_uint(z1);
        unsigned h0 = u0 & 0xFFFF0000u,   h1 = u1 & 0xFFFF0000u;
        float l0 = z0 - __uint_as_float(h0);
        float l1 = z1 - __uint_as_float(h1);
        bH[dc].u[v] = h1 | (u0 >> 16);
        bL[dc].u[v] = (__float_as_uint(l1) & 0xFFFF0000u) | (__float_as_uint(l0) >> 16);
      }
    }

    // --- one-hot A per seg-chunk, then accumulate ---
#pragma unroll
    for (int c = 0; c < 6; ++c) {
      union Au { unsigned u[8]; v16bf v; } A;
      const int mt = c * 16 + n16;
#pragma unroll
      for (int v = 0; v < 8; ++v) {
        int j0 = (v < 4) ? 2 * v : 8 + 2 * (v - 4);
        A.u[v] = (sk[j0]     == mt ? 0x00003F80u : 0u) |
                 (sk[j0 + 1] == mt ? 0x3F800000u : 0u);
      }
#pragma unroll
      for (int dc = 0; dc < 2; ++dc) {
        acc[c][dc] = __builtin_amdgcn_wmma_f32_16x16x32_bf16(
            false, A.v, false, bH[dc].v, (short)0, acc[c][dc], false, false);
        acc[c][dc] = __builtin_amdgcn_wmma_f32_16x16x32_bf16(
            false, A.v, false, bL[dc].v, (short)0, acc[c][dc], false, false);
      }
    }

    __syncthreads();
    if (i + 2 < TPB_T)
      stage_tile(z, t0 + (unsigned)(i + 2) * GRID_S, zbuf + (i & 1) * 8192, tid);
  }

  // merge per-wave WMMA accumulators into block bins, then into global
#pragma unroll
  for (int c = 0; c < 6; ++c)
#pragma unroll
    for (int dc = 0; dc < 2; ++dc)
#pragma unroll
      for (int r = 0; r < 8; ++r) {
        int M = c * 16 + half * 8 + r;
        if (M < 81) atomicAdd(&binsS[M * 32 + dc * 16 + n16], acc[c][dc][r]);
      }
  __syncthreads();
  float* gS = ws + WS_SUMS4;
  float* gC = ws + WS_CNT4;
  for (int i = tid; i < 2592; i += 256) {
    float v = binsS[i];
    if (v != 0.0f) unsafeAtomicAdd(&gS[i], v);
  }
  for (int i = tid; i < 81; i += 256) {
    float v = cntB[i];
    if (v != 0.0f) unsafeAtomicAdd(&gC[i], v);
  }
}

// ---------------- kernel 2: aggregate levels, Frechet-iterate centroids ----------------
__device__ __forceinline__ void proj32(float* v) {
  float nn = 0.0f;
#pragma unroll
  for (int d = 0; d < DIMS; ++d) nn = fmaf(v[d], v[d], nn);
  float nr = sqrtf(nn);
  if (nr > 0.95f) {
    float sc = 0.95f / (nr + 1e-12f);
#pragma unroll
    for (int d = 0; d < DIMS; ++d) v[d] *= sc;
  }
}

__global__ void k2_cent(float* ws) {
  const int t = threadIdx.x;
  if (t >= 120) return;
  const float* gS = ws + WS_SUMS4;
  const float* gC = ws + WS_CNT4;
  float* gCent = ws + WS_CENT;
  float* gNy   = ws + WS_NY;
  float* gCnt  = ws + WS_CNTA;

  int cs, n;                                  // child range in fine segments
  if (t < 3)       { cs = t * 27;        n = 27; }
  else if (t < 12) { cs = (t - 3) * 9;   n = 9;  }
  else if (t < 39) { cs = (t - 12) * 3;  n = 3;  }
  else             { cs = (t - 39);      n = 1;  }

  float cnt = 0.0f;
  for (int j = 0; j < n; ++j) cnt += gC[cs + j];

  float eu[DIMS];
#pragma unroll
  for (int d = 0; d < DIMS; ++d) {
    float sm = 0.0f;
    for (int j = 0; j < n; ++j) sm += gS[(cs + j) * DIMS + d];
    eu[d] = sm;
  }
  float safe = fmaxf(cnt, 1.0f);
#pragma unroll
  for (int d = 0; d < DIMS; ++d) eu[d] /= safe;

  float m[DIMS];
#pragma unroll
  for (int d = 0; d < DIMS; ++d) m[d] = eu[d];
  proj32(m);
  for (int it = 0; it < 10; ++it) {
#pragma unroll
    for (int d = 0; d < DIMS; ++d) m[d] = m[d] + 0.1f * (eu[d] - m[d]);
    proj32(m);
  }
  float ny = 0.0f;
#pragma unroll
  for (int d = 0; d < DIMS; ++d) {
    gCent[t * DIMS + d] = m[d];
    ny = fmaf(m[d], m[d], ny);
  }
  gNy[t]  = ny;
  gCnt[t] = cnt;
}

// ---------------- kernel 3: Poincare distances + per-(level,seg) sums ----------------
__global__ __launch_bounds__(256) void k3_dist(const float* __restrict__ z,
                                               const unsigned* __restrict__ bidx2,
                                               float* __restrict__ ws) {
  extern __shared__ float smem[];
  float* zbuf  = smem;                 // 16384 (double buffer)
  float* centL = smem + 16384;         // 120*32 = 3840
  float* nyL   = centL + 3840;         // 128 (120 used)
  float* distB = nyL + 128;            // 128 (120 used)
  // total 20480 floats = 81920 bytes

  const int tid = threadIdx.x;
  const float* gCent = ws + WS_CENT;
  const float* gNy   = ws + WS_NY;
  for (int i = tid; i < 3840; i += 256) centL[i] = gCent[i];
  for (int i = tid; i < 120;  i += 256) nyL[i]   = gNy[i];
  for (int i = tid; i < 128;  i += 256) distB[i] = 0.0f;

  const unsigned t0 = blockIdx.x;
  stage_tile(z, t0 + 0u * GRID_S, zbuf, tid);
  stage_tile(z, t0 + 1u * GRID_S, zbuf + 8192, tid);

#pragma unroll 1               // no outer unroll: compact code, low pressure
  for (int i = 0; i < TPB_T; ++i) {
    if (i + 1 < TPB_T) wait_async<8>(); else wait_async<0>();
    __syncthreads();

    const float* row = zbuf + (i & 1) * 8192 + tid * DIMS;
    const unsigned tile = t0 + (unsigned)i * GRID_S;
    const unsigned p = tile * TILEPTS + (unsigned)tid;
    const unsigned s4 = bidx2[2u * p] / 243u;
    const int r1 = (int)(s4 / 27u);
    const int r2 = 3 + (int)(s4 / 9u);
    const int r3 = 12 + (int)(s4 / 3u);
    const int r4 = 39 + (int)s4;

    float nx = 0.f, d1 = 0.f, d2 = 0.f, d3 = 0.f, d4 = 0.f;
#pragma unroll
    for (int t = 0; t < 32; ++t) {
      int d = (tid + t) & 31;                 // rotated: bank-conflict-free
      float v = row[d];
      nx = fmaf(v, v, nx);
      d1 = fmaf(v, centL[r1 * DIMS + d], d1);
      d2 = fmaf(v, centL[r2 * DIMS + d], d2);
      d3 = fmaf(v, centL[r3 * DIMS + d], d3);
      d4 = fmaf(v, centL[r4 * DIMS + d], d4);
    }
    float nrm = sqrtf(nx);
    float s = (nrm > 0.95f) ? 0.95f / (nrm + 1e-12f) : 1.0f;
    float sn = s * s * nx;                    // ||x_hyp||^2

    const int rr[4] = {r1, r2, r3, r4};
    const float dd[4] = {d1, d2, d3, d4};
#pragma unroll
    for (int l = 0; l < 4; ++l) {
      float ny = nyL[rr[l]];
      float sq = fmaxf(sn - 2.0f * s * dd[l] + ny, 0.0f);   // ||x - c||^2
      float den = fmaxf((1.0f - sn) * (1.0f - ny), 1e-12f);
      float arg = fmaxf(1.0f + 2.0f * sq / den, 1.0f + 1e-7f);
      atomicAdd(&distB[rr[l]], acoshf(arg));
    }

    __syncthreads();
    if (i + 2 < TPB_T)
      stage_tile(z, t0 + (unsigned)(i + 2) * GRID_S, zbuf + (i & 1) * 8192, tid);
  }

  __syncthreads();
  float* gD = ws + WS_DIST;
  for (int i = tid; i < 120; i += 256) {
    float v = distB[i];
    if (v != 0.0f) unsafeAtomicAdd(&gD[i], v);
  }
}

// ---------------- kernel 4: final 120-element reduction ----------------
__global__ void k4_final(const float* __restrict__ ws,
                         const float* __restrict__ lw, float* __restrict__ out) {
  if (threadIdx.x != 0) return;
  const float* gD   = ws + WS_DIST;
  const float* gCnt = ws + WS_CNTA;
  const int off[5] = {0, 3, 12, 39, 120};
  float total = 0.0f;
  for (int l = 0; l < 4; ++l) {
    float ll = 0.0f, nu = 0.0f;
    for (int t = off[l]; t < off[l + 1]; ++t) {
      float c = gCnt[t];
      if (c > 0.0f)  nu += 1.0f;
      if (c >= 2.0f) ll += gD[t] / fmaxf(c, 1.0f);
    }
    total += lw[l] * ll / fmaxf(nu, 1.0f);
  }
  out[0] = total;
}

// ---------------- host launcher ----------------
extern "C" void kernel_launch(void* const* d_in, const int* in_sizes, int n_in,
                              void* d_out, int out_size, void* d_ws, size_t ws_size,
                              hipStream_t stream) {
  const float*    z  = (const float*)d_in[0];
  const unsigned* bi = (const unsigned*)d_in[1];   // int64 indices, low word used
  const float*    lw = (const float*)d_in[2];
  float* out = (float*)d_out;
  float* ws  = (float*)d_ws;

  k0_init<<<1, 256, 0, stream>>>(ws);
  k1_sums<<<GRID_S, 256, 19584 * sizeof(float), stream>>>(z, bi, ws);
  k2_cent<<<1, 128, 0, stream>>>(ws);
  k3_dist<<<GRID_S, 256, 20480 * sizeof(float), stream>>>(z, bi, ws);
  k4_final<<<1, 32, 0, stream>>>(ws, lw, out);
}
